// Trans_3015067042537
// MI455X (gfx1250) — matrix operand order repaired and checked by reading, not compiled, not run on hardware
//
#include <hip/hip_runtime.h>

typedef __attribute__((ext_vector_type(16))) _Float16 v16h;
typedef __attribute__((ext_vector_type(8)))  float    v8f;

#define B_    4
#define C_    256
#define H_    64
#define W_    64
#define KK_   9
#define OCOFF 18
#define KTOT  2304     // 9*256
#define NCHUNK 72      // KTOT/32
#define MTILE 32       // pixels per block (two 16-row A fragments)

// ---------------- Stage 1: offset conv, exact f32 (feeds floor(), keep exact) ----
__global__ void offset_conv_kernel(const float* __restrict__ x,
                                   const float* __restrict__ w_off,
                                   float* __restrict__ offs) {
  int bid = blockIdx.x;              // b*(18*64) + ch*64 + h
  int h   = bid & 63;
  int ch  = (bid >> 6) % OCOFF;
  int b   = bid / (OCOFF * 64);
  int w   = threadIdx.x;             // 64 threads = one output row
  const float* xb = x + b * C_ * H_ * W_;
  const float* wb = w_off + ch * C_ * 9;
  float acc = 0.f;
  for (int c = 0; c < C_; ++c) {
    const float* xc = xb + c * H_ * W_;
    const float* wc = wb + c * 9;            // block-uniform -> scalar loads
#pragma unroll
    for (int ky = 0; ky < 3; ++ky) {
      int y = h - 1 + ky;
      if ((unsigned)y < 64u) {
#pragma unroll
        for (int kx = 0; kx < 3; ++kx) {
          int xx = w - 1 + kx;
          if ((unsigned)xx < 64u)
            acc = fmaf(xc[y * 64 + xx], wc[ky * 3 + kx], acc);
        }
      }
    }
  }
  offs[((b * OCOFF + ch) * 64 + h) * 64 + w] = acc;
}

// ---------------- Stage 2: pack w_def into f16 WMMA B-fragment order -------------
// Fragment f = ck*16 + nb (ck: K-chunk 0..71, nb: 16-wide N-subtile 0..15).
// Lane L (0..31), element i (0..15): N = nb*16 + (L&15), Klocal = ((L>>4)<<4) + i.
__global__ void wdef_pack_kernel(const float* __restrict__ w_def,
                                 _Float16* __restrict__ Bws) {
  int g  = blockIdx.x * 256 + threadIdx.x;   // exactly 589824 threads
  int i  = g & 15;
  int L  = (g >> 4) & 31;
  int f  = g >> 9;
  int nb = f & 15;
  int ck = f >> 4;
  int kl = ((L >> 4) << 4) + i;
  int t  = ck * 32 + kl;                     // global K index: t = kk*256 + c
  int kk = t >> 8;
  int c  = t & 255;
  int n  = nb * 16 + (L & 15);
  Bws[g] = (_Float16)w_def[(n * C_ + c) * 9 + kk];
}

// ---------------- Stage 3: deformable sampling + WMMA GEMM ----------------------
__global__ void __launch_bounds__(256)
deform_wmma_kernel(const float* __restrict__ x,
                   const float* __restrict__ offs,
                   const _Float16* __restrict__ Bws,
                   float* __restrict__ out) {
  __shared__ _Float16 sA[2][32 * 16];        // two A fragments (2 KB)
  __shared__ float    bwgt[KK_][MTILE][4];   // validity-masked bilinear weights
  __shared__ int      bidx[KK_][MTILE][4];   // clamped yc0,xc0,yc1,xc1

  int tid  = threadIdx.x;
  int lane = tid & 31;
  int wave = tid >> 5;

  int p0 = blockIdx.x * MTILE;               // 32 consecutive pixels, same (b,h)
  int b  = p0 >> 12;
  int h  = (p0 >> 6) & 63;
  int w0 = p0 & 63;                          // 0 or 32

  // Precompute masked bilinear weights + clamped coords: 9 taps x 32 pixels.
  // Validity depends only on (tap,pixel) -> fold masks into the weights so the
  // per-channel gather loop is fully branchless.
  for (int t = tid; t < KK_ * MTILE; t += 256) {
    int kk = t >> 5;
    int m  = t & 31;
    int w  = w0 + m;
    float offy = offs[((b * OCOFF + kk * 2    ) * 64 + h) * 64 + w];
    float offx = offs[((b * OCOFF + kk * 2 + 1) * 64 + h) * 64 + w];
    float py = offy + (float)(h - 1 + kk / 3);
    float px = offx + (float)(w - 1 + kk % 3);
    float y0f = floorf(py), x0f = floorf(px);
    float wy1 = py - y0f,  wx1 = px - x0f;
    float wy0 = 1.f - wy1, wx0 = 1.f - wx1;
    int y0 = (int)y0f, x0 = (int)x0f;
    int y1 = y0 + 1,   x1 = x0 + 1;
    float vy0 = ((unsigned)y0 < 64u) ? 1.f : 0.f;
    float vy1 = ((unsigned)y1 < 64u) ? 1.f : 0.f;
    float vx0 = ((unsigned)x0 < 64u) ? 1.f : 0.f;
    float vx1 = ((unsigned)x1 < 64u) ? 1.f : 0.f;
    bwgt[kk][m][0] = wy0 * wx0 * vy0 * vx0;
    bwgt[kk][m][1] = wy0 * wx1 * vy0 * vx1;
    bwgt[kk][m][2] = wy1 * wx0 * vy1 * vx0;
    bwgt[kk][m][3] = wy1 * wx1 * vy1 * vx1;
    bidx[kk][m][0] = min(max(y0, 0), 63);
    bidx[kk][m][1] = min(max(x0, 0), 63);
    bidx[kk][m][2] = min(max(y1, 0), 63);
    bidx[kk][m][3] = min(max(x1, 0), 63);
  }
  __syncthreads();

  v8f acc00 = {}, acc01 = {}, acc10 = {}, acc11 = {};
  int nb0 = wave * 2;                        // each wave owns N-subtiles nb0,nb0+1
  const float* xb = x + b * C_ * H_ * W_;

  for (int ck = 0; ck < NCHUNK; ++ck) {
    int kk = (ck * 32) >> 8;                 // tap for this chunk
    int c0 = (ck * 32) & 255;                // first channel of this chunk

    // Build both A fragments: 1024 f16 elements, 4 per thread
    // (same pixel/tap, 4 adjacent channels -> bilinear data loaded once).
    {
      int e0  = tid * 4;
      int m32 = e0 >> 5;                     // pixel 0..31
      int mt  = m32 >> 4;                    // fragment 0/1
      int m   = m32 & 15;
      int klb = e0 & 31;                     // multiple of 4
      float w00 = bwgt[kk][m32][0], w01 = bwgt[kk][m32][1];
      float w10 = bwgt[kk][m32][2], w11 = bwgt[kk][m32][3];
      int a00 = bidx[kk][m32][0] * 64 + bidx[kk][m32][1];
      int a01 = bidx[kk][m32][0] * 64 + bidx[kk][m32][3];
      int a10 = bidx[kk][m32][2] * 64 + bidx[kk][m32][1];
      int a11 = bidx[kk][m32][2] * 64 + bidx[kk][m32][3];
#pragma unroll
      for (int rep = 0; rep < 4; ++rep) {
        int kl = klb + rep;
        const float* xc = xb + (c0 + kl) * (H_ * W_);
        float val = w00 * xc[a00] + w01 * xc[a01] + w10 * xc[a10] + w11 * xc[a11];
        // A layout (matches packed B): lanes<16 hold K{0-15}, lanes>=16 hold K{16-31}
        int hi = kl >> 4;
        int L  = m + (hi ? 16 : 0);
        int i  = kl & 15;
        sA[mt][L * 16 + i] = (_Float16)val;
      }
    }
    __syncthreads();

    v16h a0 = *reinterpret_cast<const v16h*>(&sA[0][lane * 16]);
    v16h a1 = *reinterpret_cast<const v16h*>(&sA[1][lane * 16]);
    const v16h* bp = reinterpret_cast<const v16h*>(Bws) + ((size_t)(ck * 16 + nb0) * 32 + lane);
    v16h bf0 = bp[0];
    v16h bf1 = bp[32];
    acc00 = __builtin_amdgcn_wmma_f32_16x16x32_f16(false, a0, false, bf0, (short)0, acc00, false, false);
    acc10 = __builtin_amdgcn_wmma_f32_16x16x32_f16(false, a1, false, bf0, (short)0, acc10, false, false);
    acc01 = __builtin_amdgcn_wmma_f32_16x16x32_f16(false, a0, false, bf1, (short)0, acc01, false, false);
    acc11 = __builtin_amdgcn_wmma_f32_16x16x32_f16(false, a1, false, bf1, (short)0, acc11, false, false);
    __syncthreads();                          // before sA is overwritten
  }

  // Store D: lane -> N=lane&15, VGPR r -> M = r (+8 for lanes>=16); M = w offset.
  // Addresses are 32B-aligned -> float4 stores (global_store_b128).
  int ncol  = lane & 15;
  int mbase = (lane >> 4) << 3;
#pragma unroll
  for (int s = 0; s < 2; ++s) {
    int o = (nb0 + s) * 16 + ncol;
    float* obase = out + ((size_t)(b * C_ + o) * 64 + h) * 64 + w0;
#pragma unroll
    for (int mt = 0; mt < 2; ++mt) {
      v8f acc = (s == 0) ? (mt == 0 ? acc00 : acc10)
                         : (mt == 0 ? acc01 : acc11);
      float* op = obase + mt * 16 + mbase;
      float4 lo = make_float4(acc[0], acc[1], acc[2], acc[3]);
      float4 hi = make_float4(acc[4], acc[5], acc[6], acc[7]);
      *reinterpret_cast<float4*>(op)     = lo;
      *reinterpret_cast<float4*>(op + 4) = hi;
    }
  }
}

extern "C" void kernel_launch(void* const* d_in, const int* in_sizes, int n_in,
                              void* d_out, int out_size, void* d_ws, size_t ws_size,
                              hipStream_t stream) {
  const float* x     = (const float*)d_in[0];
  const float* w_off = (const float*)d_in[1];
  const float* w_def = (const float*)d_in[2];
  float* out = (float*)d_out;

  // Workspace: [offsets f32: 4*18*64*64][w_def packed f16: 72*16*32*16]
  float*    offs = (float*)d_ws;
  _Float16* Bws  = (_Float16*)((char*)d_ws + (size_t)B_ * OCOFF * H_ * W_ * sizeof(float));

  offset_conv_kernel<<<B_ * OCOFF * H_, 64, 0, stream>>>(x, w_off, offs);
  wdef_pack_kernel<<<(NCHUNK * 16 * 32 * 16) / 256, 256, 0, stream>>>(w_def, Bws);
  deform_wmma_kernel<<<(B_ * H_ * W_) / MTILE, 256, 0, stream>>>(x, offs, Bws, out);
}
